// VoxelSEResBackBone8xV1_85383949844630
// MI455X (gfx1250) — matrix operand order repaired and checked by reading, not compiled, not run on hardware
//
#include <hip/hip_runtime.h>
#include <hip/hip_bf16.h>

typedef __attribute__((ext_vector_type(16))) __bf16 v16bf;
typedef __attribute__((ext_vector_type(8)))  __bf16 v8bf;
typedef __attribute__((ext_vector_type(8)))  float  v8f;
typedef __attribute__((ext_vector_type(4)))  float  v4f;
typedef __attribute__((ext_vector_type(4)))  unsigned short v4us;

__device__ __forceinline__ unsigned short f32_to_bf16_raw(float f) {
    union { float f; unsigned int u; } v; v.f = f;
    unsigned int u = v.u;
    u += 0x7fffu + ((u >> 16) & 1u);   // round-to-nearest-even
    return (unsigned short)(u >> 16);
}

#define TM      16
#define CHUNK   32
#define ASTRIDE 40   // ushorts per LDS row (80B stride spreads banks)

// ---------------------------------------------------------------------------
// Weight repack: f32 [K,Cin,Cout] -> bf16 WMMA B-fragments, zero-padded.
// flat index: ((((kk*nch + ch)*ncc + cc)*32 + lane)*16 + e
// element e of lane (h=lane>>4, c=lane&15) = W[kk, ch*32 + 16h + e, cc*16 + c]
// ---------------------------------------------------------------------------
__global__ void repack_w_kernel(const float* __restrict__ W, int K, int Cin, int Cout,
                                unsigned short* __restrict__ Wp)
{
    int nch = (Cin + 31) / 32, ncc = Cout >> 4;
    long long total = (long long)K * nch * ncc * 512;
    long long st = (long long)gridDim.x * blockDim.x;
    for (long long i = (long long)blockIdx.x * blockDim.x + threadIdx.x; i < total; i += st) {
        int e = (int)(i & 15);
        long long r = i >> 4;
        int lane = (int)(r & 31); r >>= 5;
        int cc = (int)(r % ncc); r /= ncc;
        int ch = (int)(r % nch);
        int kk = (int)(r / nch);
        int ci = ch * 32 + 16 * (lane >> 4) + e;
        int co = cc * 16 + (lane & 15);
        float v = (ci < Cin) ? W[((long long)kk * Cin + ci) * Cout + co] : 0.f;
        Wp[i] = f32_to_bf16_raw(v);
    }
}

// ---------------------------------------------------------------------------
// Gather -> bf16 WMMA GEMM -> (bias)*scale+offset -> (relu)
// ---------------------------------------------------------------------------
template<int CIN, int COUT>
__global__ __launch_bounds__(128, 2)
void gconv_wmma_kernel(const float* __restrict__ fin,
                       const int* __restrict__ rb, int K, int Nout,
                       const unsigned short* __restrict__ Wp,
                       const float* __restrict__ bias,
                       const float* __restrict__ sc,
                       const float* __restrict__ ofs,
                       int do_relu,
                       float* __restrict__ out)
{
    constexpr int NCH = (CIN + 31) / 32;   // 32-wide contraction chunks
    constexpr int NCC = COUT / 16;         // 16-wide column chunks (1..8)

    __shared__ __align__(16) unsigned short atile[2][TM * ASTRIDE];

    const int tile0 = blockIdx.x * TM;
    const int lane  = threadIdx.x & 31;
    const int wave  = threadIdx.x >> 5;    // 0..3
    const int col   = lane & 15;
    const int h     = lane >> 4;

    // filler: thread owns (row frow, 4 columns at fcol) of the 16x32 A tile
    const int frow = threadIdx.x >> 3;
    const int fcol = (threadIdx.x & 7) * 4;
    const int fn   = tile0 + frow;

    v8f acc0 = {0.f,0.f,0.f,0.f,0.f,0.f,0.f,0.f};
    v8f acc1 = acc0;

    const int S = K * NCH;

    auto fill = [&](int t, int buf) {
        int kk  = t / NCH;
        int ci0 = (t - kk * NCH) * CHUNK;
        int id  = (fn < Nout) ? rb[(long long)fn * K + kk] : -1;
        float v0 = 0.f, v1 = 0.f, v2 = 0.f, v3 = 0.f;
        if (id >= 0) {
            const float* p = fin + (long long)id * CIN + ci0 + fcol;
            if constexpr (CIN % CHUNK == 0) {
                v4f f = *(const v4f*)p;          // 16B-aligned: CIN%4==0, fcol%4==0
                v0 = f.x; v1 = f.y; v2 = f.z; v3 = f.w;
            } else {
                int ci = ci0 + fcol;
                if (ci + 0 < CIN) v0 = p[0];
                if (ci + 1 < CIN) v1 = p[1];
                if (ci + 2 < CIN) v2 = p[2];
                if (ci + 3 < CIN) v3 = p[3];
            }
        }
        v4us pk = { f32_to_bf16_raw(v0), f32_to_bf16_raw(v1),
                    f32_to_bf16_raw(v2), f32_to_bf16_raw(v3) };
        *(v4us*)&atile[buf][frow * ASTRIDE + fcol] = pk;
    };

    auto compute = [&](int t, int buf) {
        // A fragment: lanes<16 hold K {0..15}, lanes>=16 hold K {16..31}
        const unsigned short* arow = &atile[buf][col * ASTRIDE];
        v8bf alo = *(const v8bf*)(arow + 16 * h);
        v8bf ahi = *(const v8bf*)(arow + 16 * h + 8);
        v16bf afrag = __builtin_shufflevector(alo, ahi,
            0,1,2,3,4,5,6,7,8,9,10,11,12,13,14,15);

        const unsigned short* wb = Wp + (long long)t * NCC * 512;
        __builtin_prefetch(wb + 2 * NCC * 512, 0, 0);   // stream next panels into L2

        if (wave < NCC) {
            const v8bf* bp = (const v8bf*)(wb + (wave * 32 + lane) * 16);
            v16bf bfrag = __builtin_shufflevector(bp[0], bp[1],
                0,1,2,3,4,5,6,7,8,9,10,11,12,13,14,15);
            acc0 = __builtin_amdgcn_wmma_f32_16x16x32_bf16(
                false, afrag, false, bfrag, (short)0, acc0, false, false);
        }
        if constexpr (NCC > 4) {
            const v8bf* bp = (const v8bf*)(wb + ((wave + 4) * 32 + lane) * 16);
            v16bf bfrag = __builtin_shufflevector(bp[0], bp[1],
                0,1,2,3,4,5,6,7,8,9,10,11,12,13,14,15);
            acc1 = __builtin_amdgcn_wmma_f32_16x16x32_bf16(
                false, afrag, false, bfrag, (short)0, acc1, false, false);
        }
    };

    fill(0, 0);
    __syncthreads();
    for (int t = 0; t < S; ++t) {
        int buf = t & 1;
        compute(t, buf);                 // WMMA on buf
        if (t + 1 < S) fill(t + 1, buf ^ 1);   // overlap next gather with WMMA
        __syncthreads();
    }

    // epilogue; C/D layout: VGPR v -> row v + 8*h, column lane&15
    #pragma unroll
    for (int which = 0; which < 2; ++which) {
        int cc = which ? (wave + 4) : wave;
        if (which == 1 && NCC <= 4) continue;
        if (cc >= NCC) continue;
        v8f& acc = which ? acc1 : acc0;
        int co = cc * 16 + col;
        float b = bias ? bias[co] : 0.f;
        float s = sc[co], o = ofs[co];
        #pragma unroll
        for (int v = 0; v < 8; ++v) {
            int rg = tile0 + v + 8 * h;
            if (rg < Nout) {
                float val = (acc[v] + b) * s + o;
                if (do_relu) val = fmaxf(val, 0.f);
                out[(long long)rg * COUT + co] = val;
            }
        }
    }
}

// ---------------------------------------------------------------------------
// SE helpers / elementwise
// ---------------------------------------------------------------------------
__global__ void zero_kernel(float* __restrict__ p, long long n) {
    long long i = (long long)blockIdx.x * blockDim.x + threadIdx.x;
    long long st = (long long)gridDim.x * blockDim.x;
    for (; i < n; i += st) p[i] = 0.f;
}

__global__ void seg_sum_kernel(const float* __restrict__ x, const int* __restrict__ bid,
                               int N, int C, float* __restrict__ acc)  // acc: [2*C sums][2 cnt]
{
    __shared__ float ls[2 * 128 + 2];
    int tot = 2 * C + 2;
    for (int i = threadIdx.x; i < tot; i += blockDim.x) ls[i] = 0.f;
    __syncthreads();
    long long total = (long long)N * C;
    long long st = (long long)gridDim.x * blockDim.x;
    for (long long i = (long long)blockIdx.x * blockDim.x + threadIdx.x; i < total; i += st) {
        int r = (int)(i / C), c = (int)(i - (long long)r * C);
        int b = bid[r];
        atomicAdd(&ls[b * C + c], x[i]);
        if (c == 0) atomicAdd(&ls[2 * C + b], 1.f);
    }
    __syncthreads();
    for (int i = threadIdx.x; i < tot; i += blockDim.x) atomicAdd(&acc[i], ls[i]);
}

__global__ void se_mlp_kernel(const float* __restrict__ acc,
                              const float* __restrict__ w1, const float* __restrict__ w2,
                              int C, int hdim, float* __restrict__ y)
{
    __shared__ float mean[256];
    __shared__ float hid[16];
    int tid = threadIdx.x;
    for (int i = tid; i < 2 * C; i += blockDim.x) {
        float cnt = acc[2 * C + (i / C)];
        mean[i] = acc[i] / fmaxf(cnt, 1.f);
    }
    __syncthreads();
    if (tid < 2 * hdim) {
        int b = tid / hdim, j = tid - b * hdim;
        float s = 0.f;
        for (int c = 0; c < C; ++c) s += mean[b * C + c] * w1[j * C + c];
        hid[tid] = fmaxf(s, 0.f);
    }
    __syncthreads();
    for (int i = tid; i < 2 * C; i += blockDim.x) {
        int b = i / C, c = i - b * C;
        float s = 0.f;
        for (int j = 0; j < hdim; ++j) s += hid[b * hdim + j] * w2[c * hdim + j];
        y[i] = 1.f / (1.f + __expf(-s));
    }
}

__global__ void se_apply_kernel(const float* __restrict__ o2, const float* __restrict__ y,
                                const int* __restrict__ bid, const float* __restrict__ resid,
                                float* __restrict__ out, int N, int C)
{
    long long total = (long long)N * C;
    long long st = (long long)gridDim.x * blockDim.x;
    for (long long i = (long long)blockIdx.x * blockDim.x + threadIdx.x; i < total; i += st) {
        int r = (int)(i / C), c = (int)(i - (long long)r * C);
        out[i] = fmaxf(o2[i] * y[bid[r] * C + c] + resid[i], 0.f);
    }
}

__global__ void scatter_add_kernel(const float* __restrict__ src, const int* __restrict__ map,
                                   float* __restrict__ dst, int M, int C)
{
    long long total = (long long)M * C;
    long long st = (long long)gridDim.x * blockDim.x;
    for (long long i = (long long)blockIdx.x * blockDim.x + threadIdx.x; i < total; i += st) {
        int r = (int)(i / C), c = (int)(i - (long long)r * C);
        dst[(long long)map[r] * C + c] += src[i];
    }
}

// ---------------------------------------------------------------------------
// Host orchestration
// ---------------------------------------------------------------------------
extern "C" void kernel_launch(void* const* d_in, const int* in_sizes, int n_in,
                              void* d_out, int out_size, void* d_ws, size_t ws_size,
                              hipStream_t stream)
{
    auto F = [&](int i) { return (const float*)d_in[i]; };
    auto I = [&](int i) { return (const int*)d_in[i]; };

    const int VF = 0;
    const int RB1 = 123, RB2 = 124, RB3 = 125, RB4 = 126;
    const int D2 = 127, D3 = 128, D4 = 129, D6 = 130, DOUT = 131;
    const int MA = 132, MB = 133;
    const int B1 = 134, B2 = 135, B3 = 136, B4 = 137, BU = 138;

    const int N1  = in_sizes[RB1] / 27;
    const int N2  = in_sizes[RB2] / 27;
    const int N3  = in_sizes[RB3] / 27;
    const int N4  = in_sizes[RB4] / 27;
    const int N4b = in_sizes[D6]  / 27;
    const int Nu  = in_sizes[BU];
    const int NO  = in_sizes[DOUT] / 3;

    int n4max = N4 > N4b ? N4 : N4b;
    if (Nu > n4max) n4max = Nu;

    float* ws = (float*)d_ws;
    size_t off = 0;
    auto alloc = [&](size_t n) { float* p = ws + off; off += n; return p; };
    float* f1a = alloc((size_t)N1 * 16); float* f1b = alloc((size_t)N1 * 16); float* f1c = alloc((size_t)N1 * 16);
    float* f2a = alloc((size_t)N2 * 32); float* f2b = alloc((size_t)N2 * 32); float* f2c = alloc((size_t)N2 * 32);
    float* f3a = alloc((size_t)N3 * 64); float* f3b = alloc((size_t)N3 * 64); float* f3c = alloc((size_t)N3 * 64);
    float* f4a = alloc((size_t)n4max * 128); float* f4b = alloc((size_t)n4max * 128); float* f4c = alloc((size_t)n4max * 128);
    float* se_acc = alloc(2 * 128 + 2);
    float* se_y   = alloc(2 * 128);
    // packed-weight scratch: max conv = 27 taps * 128 (padded Cin) * 128 Cout bf16
    unsigned short* wpack = (unsigned short*)alloc((size_t)27 * 128 * 128 / 2 + 64);

    auto ew_grid = [&](long long total) {
        long long g = (total + 255) / 256;
        if (g > 2048) g = 2048;
        if (g < 1) g = 1;
        return dim3((unsigned)g);
    };

    auto gconv = [&](const float* fin, int Cin, const int* rbp, int K, int Nr,
                     int wi, const float* bias, int relu, float* outp) {
        int Cout = in_sizes[wi + 1];          // scale vector length == Cout
        int nch = (Cin + 31) / 32, ncc = Cout / 16;
        long long wtot = (long long)K * nch * ncc * 512;
        repack_w_kernel<<<ew_grid(wtot), dim3(256), 0, stream>>>(F(wi), K, Cin, Cout, wpack);

        dim3 g((unsigned)((Nr + TM - 1) / TM)), b(128);
        const float* s = F(wi + 1); const float* o = F(wi + 2);
        if      (Cin == 5   && Cout == 16 ) gconv_wmma_kernel<5,  16 ><<<g,b,0,stream>>>(fin, rbp, K, Nr, wpack, bias, s, o, relu, outp);
        else if (Cin == 16  && Cout == 16 ) gconv_wmma_kernel<16, 16 ><<<g,b,0,stream>>>(fin, rbp, K, Nr, wpack, bias, s, o, relu, outp);
        else if (Cin == 16  && Cout == 32 ) gconv_wmma_kernel<16, 32 ><<<g,b,0,stream>>>(fin, rbp, K, Nr, wpack, bias, s, o, relu, outp);
        else if (Cin == 32  && Cout == 32 ) gconv_wmma_kernel<32, 32 ><<<g,b,0,stream>>>(fin, rbp, K, Nr, wpack, bias, s, o, relu, outp);
        else if (Cin == 32  && Cout == 64 ) gconv_wmma_kernel<32, 64 ><<<g,b,0,stream>>>(fin, rbp, K, Nr, wpack, bias, s, o, relu, outp);
        else if (Cin == 64  && Cout == 64 ) gconv_wmma_kernel<64, 64 ><<<g,b,0,stream>>>(fin, rbp, K, Nr, wpack, bias, s, o, relu, outp);
        else if (Cin == 64  && Cout == 128) gconv_wmma_kernel<64, 128><<<g,b,0,stream>>>(fin, rbp, K, Nr, wpack, bias, s, o, relu, outp);
        else if (Cin == 128 && Cout == 128) gconv_wmma_kernel<128,128><<<g,b,0,stream>>>(fin, rbp, K, Nr, wpack, bias, s, o, relu, outp);
    };

    auto block = [&](float* x, const int* rbp, int Nr, const int* bidp, int C, int base,
                     float* t1, float* t2, float* outp) {
        gconv(x,  C, rbp, 27, Nr, base + 0, F(base + 3), 1, t1);   // conv1 + bias + relu
        gconv(t1, C, rbp, 27, Nr, base + 4, F(base + 7), 0, t2);   // conv2 + bias (no relu)
        zero_kernel<<<dim3(1), dim3(256), 0, stream>>>(se_acc, (long long)(2 * C + 2));
        seg_sum_kernel<<<ew_grid((long long)Nr * C), dim3(256), 0, stream>>>(t2, bidp, Nr, C, se_acc);
        int hdim = C / 16; if (hdim < 1) hdim = 1;
        se_mlp_kernel<<<dim3(1), dim3(256), 0, stream>>>(se_acc, F(base + 8), F(base + 9), C, hdim, se_y);
        se_apply_kernel<<<ew_grid((long long)Nr * C), dim3(256), 0, stream>>>(t2, se_y, bidp, x, outp, Nr, C);
    };

    // ---- level 1 (C=16) ----
    gconv(F(VF), 5, I(RB1), 27, N1, 3, nullptr, 1, f1a);            // in0
    block(f1a, I(RB1), N1, I(B1), 16, 6,  f1b, f1c, f1b);           // blkA
    block(f1b, I(RB1), N1, I(B1), 16, 16, f1c, f1a, f1c);           // blkB
    // ---- level 2 (C=32) ----
    gconv(f1c, 16, I(D2), 27, N2, 26, nullptr, 1, f2a);             // down2
    block(f2a, I(RB2), N2, I(B2), 32, 29, f2b, f2c, f2b);           // blkC
    block(f2b, I(RB2), N2, I(B2), 32, 39, f2c, f2a, f2c);           // blkD
    // ---- level 3 (C=64) ----
    gconv(f2c, 32, I(D3), 27, N3, 49, nullptr, 1, f3a);             // down3
    block(f3a, I(RB3), N3, I(B3), 64, 52, f3b, f3c, f3b);           // blkE
    block(f3b, I(RB3), N3, I(B3), 64, 62, f3c, f3a, f3c);           // blkF
    block(f3c, I(RB3), N3, I(B3), 64, 72, f3a, f3b, f3a);           // blkG -> p1 = f3a
    // ---- level 4 branch ----
    gconv(f3a, 64, I(D4), 27, N4, 82, nullptr, 1, f4a);             // down4 -> p2 = f4a
    gconv(f3a, 64, I(RB3), 27, N3, 85, nullptr, 1, f3b);            // t11  -> p1 = f3b
    block(f3b, I(RB3), N3, I(B3), 64, 88, f3c, f3a, f3c);           // blkH -> p1 = f3c
    block(f4a, I(RB4), N4, I(B4), 128, 98, f4b, f4c, f4b);          // blkI -> p2 = f4b
    gconv(f3c, 64,  I(RB3), 27, N3, 108, nullptr, 1, f3a);          // p1c  -> p1o = f3a
    gconv(f4b, 128, I(RB4), 27, N4, 111, nullptr, 1, f4c);          // p2c  -> p2o = f4c
    gconv(f3a, 64,  I(D6),  27, N4b, 114, nullptr, 1, f4a);         // p1t2 -> a = f4a
    gconv(f4c, 128, I(RB4), 27, N4, 117, nullptr, 1, f4b);          // p2t2 -> b = f4b
    // ---- sparse_add coalesce into u = f4c ----
    zero_kernel<<<ew_grid((long long)Nu * 128), dim3(256), 0, stream>>>(f4c, (long long)Nu * 128);
    scatter_add_kernel<<<ew_grid((long long)N4b * 128), dim3(256), 0, stream>>>(f4a, I(MB), f4c, N4b, 128);
    scatter_add_kernel<<<ew_grid((long long)N4  * 128), dim3(256), 0, stream>>>(f4b, I(MA), f4c, N4, 128);
    // ---- output conv (kernel=3, Cin=128->Cout=128) ----
    gconv(f4c, 128, I(DOUT), 3, NO, 120, nullptr, 1, (float*)d_out);
}
